// EnhancedUnderstandingNet_66529043415109
// MI455X (gfx1250) — matrix operand order repaired and checked by reading, not compiled, not run on hardware
//
#include <hip/hip_runtime.h>
#include <hip/hip_bf16.h>
#include <math.h>

#define BATCH   2048
#define DMODEL  1024
#define NSCHEMA 32768
#define TOPK    4
#define NSPLIT  8
#define EPSF    1e-8f
#define NEG_BIG (-3.0e38f)

typedef __attribute__((ext_vector_type(16))) __bf16 v16bf;
typedef __attribute__((ext_vector_type(8)))  float  v8f;
typedef __attribute__((ext_vector_type(4)))  int    v4i;

#if __has_builtin(__builtin_amdgcn_global_load_async_to_lds_b128) && \
    __has_builtin(__builtin_amdgcn_s_wait_asynccnt)
#define USE_ASYNC_LDS 1
#define AS1 __attribute__((address_space(1)))
#define AS3 __attribute__((address_space(3)))
#else
#define USE_ASYNC_LDS 0
#endif

union BF16Frag {
    v16bf          v;
    unsigned short u[16];
    uint4          q[2];
};

__device__ __forceinline__ unsigned short bf16_bits(float f) {
    union { __bf16 b; unsigned short u; } c;
    c.b = (__bf16)f;                      // RNE convert -> v_cvt_pk_bf16_f32 path
    return c.u;
}

__device__ __forceinline__ void top4_insert(float s, int j, float tv[TOPK], int ti[TOPK]) {
    if (s > tv[TOPK - 1]) {
        tv[TOPK - 1] = s; ti[TOPK - 1] = j;
#pragma unroll
        for (int q = TOPK - 1; q > 0; --q) {
            if (tv[q] > tv[q - 1]) {
                float a = tv[q]; tv[q] = tv[q - 1]; tv[q - 1] = a;
                int   b = ti[q]; ti[q] = ti[q - 1]; ti[q - 1] = b;
            }
        }
    }
}

// ---------------- kernel 1: W f32 -> bf16 ----------------
__global__ __launch_bounds__(256) void cvt_w_kernel(const float* __restrict__ src,
                                                    unsigned short* __restrict__ dst,
                                                    int n4) {
    int i = blockIdx.x * blockDim.x + threadIdx.x;
    if (i < n4) {
        float4 v = ((const float4*)src)[i];
        dst[i * 4 + 0] = bf16_bits(v.x);
        dst[i * 4 + 1] = bf16_bits(v.y);
        dst[i * 4 + 2] = bf16_bits(v.z);
        dst[i * 4 + 3] = bf16_bits(v.w);
    }
}

// ---------------- kernel 2: keys f32 -> bf16 + row norms ----------------
__global__ __launch_bounds__(256) void prep_keys_kernel(const float* __restrict__ keys,
                                                        unsigned short* __restrict__ Kbf,
                                                        float* __restrict__ kn) {
    __shared__ float red[256];
    const int row = blockIdx.x;
    const float* src = keys + (size_t)row * DMODEL;
    unsigned short* dst = Kbf + (size_t)row * DMODEL;
    const int c = threadIdx.x * 4;                       // 256*4 == 1024 exactly
    float4 v = *(const float4*)(src + c);
    dst[c + 0] = bf16_bits(v.x);
    dst[c + 1] = bf16_bits(v.y);
    dst[c + 2] = bf16_bits(v.z);
    dst[c + 3] = bf16_bits(v.w);
    red[threadIdx.x] = v.x * v.x + v.y * v.y + v.z * v.z + v.w * v.w;
    __syncthreads();
    for (int s = 128; s > 0; s >>= 1) {
        if (threadIdx.x < s) red[threadIdx.x] += red[threadIdx.x + s];
        __syncthreads();
    }
    if (threadIdx.x == 0) kn[row] = sqrtf(red[0]);
}

// ---------------- kernel 3: zero qn2 ----------------
__global__ void zero_kernel(float* __restrict__ p, int n) {
    int i = blockIdx.x * blockDim.x + threadIdx.x;
    if (i < n) p[i] = 0.0f;
}

// ---------------- kernel 4: pq = query @ W^T + b (bf16 WMMA, f32 accum) ----------------
// 8 waves/block, each wave a 16x64 strip; A(f32->bf16) and B double-buffered so the
// next K-step's loads are in flight while the current 4 WMMAs execute.
__global__ __launch_bounds__(256) void gemm_pq_kernel(const float* __restrict__ query,
                                                      const unsigned short* __restrict__ Wbf,
                                                      const float* __restrict__ bias,
                                                      unsigned short* __restrict__ PQbf,
                                                      float* __restrict__ qn2) {
    const int lane  = threadIdx.x & 31;
    const int wave  = threadIdx.x >> 5;
    const int lm    = lane & 15;
    const int hi    = (lane >> 4) & 1;
    const int mbase = blockIdx.y * 16;
    const int nbase = blockIdx.x * 512 + wave * 64;

    const float* qrow = query + (size_t)(mbase + lm) * DMODEL + hi * 8;
    const unsigned short* wrow = Wbf + (size_t)(nbase + lm) * DMODEL + hi * 8;

    float4 fc[4];                // current A tile, raw f32
    BF16Frag bc[4];              // current B fragments
    fc[0] = *(const float4*)(qrow);
    fc[1] = *(const float4*)(qrow + 4);
    fc[2] = *(const float4*)(qrow + 16);
    fc[3] = *(const float4*)(qrow + 20);
#pragma unroll
    for (int t = 0; t < 4; ++t) {
        const unsigned short* wr = wrow + (size_t)(t * 16) * DMODEL;
        bc[t].q[0] = *(const uint4*)(wr);
        bc[t].q[1] = *(const uint4*)(wr + 16);
    }

    v8f acc[4] = {};
    for (int kb = 0; kb < DMODEL; kb += 32) {
        // issue next K-step loads first (clamped to 0 on last iter: dead but harmless)
        const int kc = (kb + 32 < DMODEL) ? (kb + 32) : 0;
        float4 fn[4];
        BF16Frag bn[4];
        fn[0] = *(const float4*)(qrow + kc);
        fn[1] = *(const float4*)(qrow + kc + 4);
        fn[2] = *(const float4*)(qrow + kc + 16);
        fn[3] = *(const float4*)(qrow + kc + 20);
#pragma unroll
        for (int t = 0; t < 4; ++t) {
            const unsigned short* wr = wrow + (size_t)(t * 16) * DMODEL + kc;
            bn[t].q[0] = *(const uint4*)(wr);
            bn[t].q[1] = *(const uint4*)(wr + 16);
        }

        // convert current A: halves 0..7 -> K=kb+hi*8+0..7, 8..15 -> +16
        BF16Frag a;
        a.u[0] = bf16_bits(fc[0].x); a.u[1] = bf16_bits(fc[0].y);
        a.u[2] = bf16_bits(fc[0].z); a.u[3] = bf16_bits(fc[0].w);
        a.u[4] = bf16_bits(fc[1].x); a.u[5] = bf16_bits(fc[1].y);
        a.u[6] = bf16_bits(fc[1].z); a.u[7] = bf16_bits(fc[1].w);
        a.u[8]  = bf16_bits(fc[2].x); a.u[9]  = bf16_bits(fc[2].y);
        a.u[10] = bf16_bits(fc[2].z); a.u[11] = bf16_bits(fc[2].w);
        a.u[12] = bf16_bits(fc[3].x); a.u[13] = bf16_bits(fc[3].y);
        a.u[14] = bf16_bits(fc[3].z); a.u[15] = bf16_bits(fc[3].w);

#pragma unroll
        for (int t = 0; t < 4; ++t)
            acc[t] = __builtin_amdgcn_wmma_f32_16x16x32_bf16(false, a.v, false, bc[t].v,
                                                             (short)0, acc[t], false, false);
#pragma unroll
        for (int t = 0; t < 4; ++t) { fc[t] = fn[t]; bc[t] = bn[t]; }
    }

#pragma unroll
    for (int t = 0; t < 4; ++t) {
        const int n = nbase + t * 16 + lm;
        const float bv = bias[n];
#pragma unroll
        for (int v = 0; v < 8; ++v) {
            const int m = mbase + v + hi * 8;
            float val = acc[t][v] + bv;
            PQbf[(size_t)m * DMODEL + n] = bf16_bits(val);
            atomicAdd(&qn2[m], val * val);
        }
    }
}

// ---------------- kernel 5: fused sims (WMMA) + masked top-4 per N-split ----------------
__global__ __launch_bounds__(256) void sims_topk_kernel(const unsigned short* __restrict__ PQbf,
                                                        const unsigned short* __restrict__ Kbf,
                                                        const float* __restrict__ qn2,
                                                        const float* __restrict__ kn,
                                                        const unsigned char* __restrict__ active,
                                                        float* __restrict__ pval,
                                                        int* __restrict__ pidx) {
    __shared__ unsigned short spq[16][DMODEL + 8];   // padded rows: bank-conflict-free b128 reads
    __shared__ float sqn[16];
    __shared__ float sc[8][16][17];
    __shared__ float mval[8][16][TOPK];
    __shared__ int   midx[8][16][TOPK];

    const int tid   = threadIdx.x;
    const int lane  = tid & 31;
    const int wave  = tid >> 5;
    const int lm    = lane & 15;
    const int hi    = (lane >> 4) & 1;
    const int mbase = blockIdx.y * 16;

    // stage this block's 16 pq rows (bf16) into LDS via async DMA when available
#if USE_ASYNC_LDS
    for (int i = tid; i < 16 * (DMODEL / 8); i += 256) {
        int r = i >> 7;        // /128
        int c = i & 127;
        __builtin_amdgcn_global_load_async_to_lds_b128(
            (AS1 v4i*)(PQbf + (size_t)(mbase + r) * DMODEL + c * 8),
            (AS3 v4i*)&spq[r][c * 8], 0, 0);
    }
    __builtin_amdgcn_s_wait_asynccnt(0);
#else
    for (int i = tid; i < 16 * (DMODEL / 8); i += 256) {
        int r = i >> 7;
        int c = i & 127;
        *(uint4*)&spq[r][c * 8] = *(const uint4*)(PQbf + (size_t)(mbase + r) * DMODEL + c * 8);
    }
#endif
    if (tid < 16) sqn[tid] = sqrtf(qn2[mbase + tid]);
    __syncthreads();

    float tv[TOPK]; int ti[TOPK];
#pragma unroll
    for (int s = 0; s < TOPK; ++s) { tv[s] = NEG_BIG; ti[s] = 0; }

    // per split: 256 tiles of 16 keys; per wave: 8 groups of 4 consecutive tiles
    const int jt0 = blockIdx.x * ((NSCHEMA / 16) / NSPLIT);
    const unsigned short* arow = &spq[lm][hi * 8];

    for (int g = 0; g < 8; ++g) {
        const int jbase = (jt0 + (g * 8 + wave) * 4) * 16;      // 64 consecutive key rows
        const unsigned short* krow = Kbf + (size_t)(jbase + lm) * DMODEL + hi * 8;
        if (g + 1 < 8)
            __builtin_prefetch(krow + (size_t)512 * DMODEL, 0, 1);   // next group, +512 rows

        // prologue: load B fragments for kb=0
        BF16Frag bc[4];
#pragma unroll
        for (int t = 0; t < 4; ++t) {
            const unsigned short* kr = krow + (size_t)(t * 16) * DMODEL;
            bc[t].q[0] = *(const uint4*)(kr);
            bc[t].q[1] = *(const uint4*)(kr + 16);
        }

        v8f acc[4] = {};
        for (int kb = 0; kb < DMODEL; kb += 32) {
            // issue next K-step B loads before computing on current fragments
            const int kc = (kb + 32 < DMODEL) ? (kb + 32) : 0;
            BF16Frag bn[4];
#pragma unroll
            for (int t = 0; t < 4; ++t) {
                const unsigned short* kr = krow + (size_t)(t * 16) * DMODEL + kc;
                bn[t].q[0] = *(const uint4*)(kr);
                bn[t].q[1] = *(const uint4*)(kr + 16);
            }
            BF16Frag a;
            a.q[0] = *(const uint4*)(arow + kb);        // K = kb+hi*8 + 0..7
            a.q[1] = *(const uint4*)(arow + kb + 16);   // K = kb+hi*8 + 16..23
#pragma unroll
            for (int t = 0; t < 4; ++t)
                acc[t] = __builtin_amdgcn_wmma_f32_16x16x32_bf16(false, a.v, false, bc[t].v,
                                                                 (short)0, acc[t], false, false);
#pragma unroll
            for (int t = 0; t < 4; ++t) bc[t] = bn[t];
        }

        // cosine normalization + active mask, one 16x16 subtile at a time
#pragma unroll
        for (int t = 0; t < 4; ++t) {
            const int jb  = jbase + t * 16;
            const int j   = jb + lm;
            const float knj = kn[j];
            const int   act = active[j] ? 1 : 0;
#pragma unroll
            for (int v = 0; v < 8; ++v) {
                const int ml = v + hi * 8;
                float denom = fmaxf(sqn[ml] * knj, EPSF);
                sc[wave][ml][lm] = act ? (acc[t][v] / denom) : 0.0f;
            }
            // lanes 0..15: per-row running top-4 (ascending index scan => stable tie-break)
            if (lane < 16) {
#pragma unroll
                for (int c = 0; c < 16; ++c)
                    top4_insert(sc[wave][lane][c], jb + c, tv, ti);
            }
        }
    }

    if (lane < 16) {
#pragma unroll
        for (int s = 0; s < TOPK; ++s) { mval[wave][lane][s] = tv[s]; midx[wave][lane][s] = ti[s]; }
    }
    __syncthreads();

    if (wave == 0 && lane < 16) {
        float fv[TOPK]; int fi[TOPK];
#pragma unroll
        for (int s = 0; s < TOPK; ++s) { fv[s] = NEG_BIG; fi[s] = 0; }
        for (int w = 0; w < 8; ++w)
#pragma unroll
            for (int s = 0; s < TOPK; ++s)
                top4_insert(mval[w][lane][s], midx[w][lane][s], fv, fi);
        const int row = mbase + lane;
        const size_t base = ((size_t)blockIdx.x * BATCH + row) * TOPK;
#pragma unroll
        for (int s = 0; s < TOPK; ++s) { pval[base + s] = fv[s]; pidx[base + s] = fi[s]; }
    }
}

// ---------------- kernel 6: merge splits, gather values, mean ----------------
__global__ __launch_bounds__(128) void merge_gather_kernel(const float* __restrict__ pval,
                                                           const int* __restrict__ pidx,
                                                           const float* __restrict__ values,
                                                           const unsigned char* __restrict__ active,
                                                           float* __restrict__ out) {
    __shared__ int   sidx[TOPK];
    __shared__ int   sact[TOPK];
    __shared__ float sinv;
    const int row = blockIdx.x;

    if (threadIdx.x == 0) {
        float fv[TOPK]; int fi[TOPK];
#pragma unroll
        for (int s = 0; s < TOPK; ++s) { fv[s] = NEG_BIG; fi[s] = 0; }
        for (int sp = 0; sp < NSPLIT; ++sp) {
            const size_t base = ((size_t)sp * BATCH + row) * TOPK;
#pragma unroll
            for (int s = 0; s < TOPK; ++s)
                top4_insert(pval[base + s], pidx[base + s], fv, fi);
        }
        int cnt = 0;
#pragma unroll
        for (int s = 0; s < TOPK; ++s) {
            int a = active[fi[s]] ? 1 : 0;
            sidx[s] = fi[s]; sact[s] = a; cnt += a;
        }
        sinv = (cnt > 0) ? (1.0f / (float)cnt) : 0.0f;
    }
    __syncthreads();

    const int d = threadIdx.x * 8;                    // 128*8 == 1024
    float accv[8] = {0, 0, 0, 0, 0, 0, 0, 0};
#pragma unroll
    for (int s = 0; s < TOPK; ++s) {
        if (sact[s]) {
            const float* vr = values + (size_t)sidx[s] * DMODEL + d;
            float4 x0 = *(const float4*)(vr);
            float4 x1 = *(const float4*)(vr + 4);
            accv[0] += x0.x; accv[1] += x0.y; accv[2] += x0.z; accv[3] += x0.w;
            accv[4] += x1.x; accv[5] += x1.y; accv[6] += x1.z; accv[7] += x1.w;
        }
    }
    const float inv = sinv;
    float4 o0 = make_float4(accv[0] * inv, accv[1] * inv, accv[2] * inv, accv[3] * inv);
    float4 o1 = make_float4(accv[4] * inv, accv[5] * inv, accv[6] * inv, accv[7] * inv);
    float* orow = out + (size_t)row * DMODEL + d;
    *(float4*)(orow)     = o0;
    *(float4*)(orow + 4) = o1;
}

// ---------------- host launcher ----------------
extern "C" void kernel_launch(void* const* d_in, const int* in_sizes, int n_in,
                              void* d_out, int out_size, void* d_ws, size_t ws_size,
                              hipStream_t stream) {
    const float* query  = (const float*)d_in[0];
    const float* W      = (const float*)d_in[1];
    const float* bias   = (const float*)d_in[2];
    const float* keys   = (const float*)d_in[3];
    const float* values = (const float*)d_in[4];
    const unsigned char* active = (const unsigned char*)d_in[5];   // jnp bool = 1 byte
    float* out = (float*)d_out;

    // workspace carve-up (256B-aligned)
    char* ws = (char*)d_ws;
    size_t off = 0;
    auto carve = [&](size_t bytes) {
        char* p = ws + off;
        off = (off + bytes + 255) & ~(size_t)255;
        return p;
    };
    unsigned short* Wbf  = (unsigned short*)carve((size_t)DMODEL * DMODEL * 2);
    unsigned short* Kbf  = (unsigned short*)carve((size_t)NSCHEMA * DMODEL * 2);
    unsigned short* PQbf = (unsigned short*)carve((size_t)BATCH * DMODEL * 2);
    float* qn2  = (float*)carve((size_t)BATCH * 4);
    float* kn   = (float*)carve((size_t)NSCHEMA * 4);
    float* pval = (float*)carve((size_t)NSPLIT * BATCH * TOPK * 4);
    int*   pidx = (int*)carve((size_t)NSPLIT * BATCH * TOPK * 4);
    (void)ws_size; (void)n_in; (void)in_sizes; (void)out_size;

    // 1. W -> bf16
    cvt_w_kernel<<<(DMODEL * DMODEL / 4 + 255) / 256, 256, 0, stream>>>(W, Wbf, DMODEL * DMODEL / 4);
    // 2. keys -> bf16 + norms
    prep_keys_kernel<<<NSCHEMA, 256, 0, stream>>>(keys, Kbf, kn);
    // 3. zero qn2
    zero_kernel<<<(BATCH + 255) / 256, 256, 0, stream>>>(qn2, BATCH);
    // 4. pq GEMM (WMMA, 16x64 per wave, double-buffered)
    gemm_pq_kernel<<<dim3(DMODEL / 512, BATCH / 16), 256, 0, stream>>>(query, Wbf, bias, PQbf, qn2);
    // 5. fused sims + top-4 (WMMA, 16x64 per wave per group, double-buffered)
    sims_topk_kernel<<<dim3(NSPLIT, BATCH / 16), 256, 0, stream>>>(PQbf, Kbf, qn2, kn, active, pval, pidx);
    // 6. merge + gather + mean
    merge_gather_kernel<<<BATCH, 128, 0, stream>>>(pval, pidx, values, active, out);
}
